// Layer_29996051595786
// MI455X (gfx1250) — matrix operand order repaired and checked by reading, not compiled, not run on hardware
//
#include <hip/hip_runtime.h>
#include <stdint.h>

// ---------------------------------------------------------------------------
// Fused Conv3x3(64->128) + BN(eval) + LIF spike scan (T=4) for MI455X/gfx1250.
// 3x3 conv = 9 shifted GEMMs (K=Cin=64), bf16 WMMA, f32 accumulate.
// Prep kernel: weights f32->bf16 into d_ws ([p][m][c], L2-resident) + BN fold.
// Main kernel: A-fragments from L2 (global_load_b128), input rows staged
// HBM->LDS once via global_load_async_to_lds_b128 (ASYNCcnt), 3 kx-pre-shifted
// bf16 copies in LDS; each B element is one 2-byte ds read at stride CSTR.
// LDS = 85 KB -> 3 WGs/WGP; LIF membrane state lives in VGPRs across T.
// ---------------------------------------------------------------------------

typedef __attribute__((ext_vector_type(16))) __bf16 v16bf;
typedef __attribute__((ext_vector_type(8)))  float  v8f;

#define T_    4
#define B_    64
#define CIN   64
#define COUT  128
#define H_    32
#define W_    32

#define CSTR  136    // channel stride (bf16 units) in shifted-x LDS; 272B (bank-decorrelating)

// LDS: 84,992 B total -> 3 workgroups per WGP (320 KB)
#define XS_BYTES   (3 * CIN * CSTR * 2)    // 52224: 3 kx-shifted bf16 input copies
#define XRAW_OFF   XS_BYTES
#define XRAW_BYTES (CIN * 4 * W_ * 4)      // 32768: raw f32 rows (async landing pad)
#define SMEM_TOTAL (XS_BYTES + XRAW_BYTES)

// d_ws layout: bf16 weights [9][128][64] (147456 B), then scale[128], shift[128] f32
#define WS_SCALE_BYTE_OFF (9 * COUT * CIN * 2)

__device__ __forceinline__ uint16_t f32_to_bf16(float f) {
    union { float f; uint32_t u; } c; c.f = f;
    uint32_t u = c.u;
    uint32_t r = u + 0x7FFFu + ((u >> 16) & 1u);   // round-to-nearest-even
    return (uint16_t)(r >> 16);
}

union AFrag { v16bf v; uint4 h[2]; };

// ---- prep: convert weights OIHW f32 -> bf16 [p][m][c]; fold bias into BN ----
__global__ __launch_bounds__(256)
void prep_kernel(const float* __restrict__ conv_w, const float* __restrict__ conv_b,
                 const float* __restrict__ bn_gamma, const float* __restrict__ bn_beta,
                 const float* __restrict__ bn_mean, const float* __restrict__ bn_var,
                 uint16_t* __restrict__ wbf, float* __restrict__ scale,
                 float* __restrict__ shift) {
    int i = blockIdx.x * 256 + threadIdx.x;
    if (i < COUT * CIN * 9) {
        int m = i / 576;
        int r = i - m * 576;
        int c = r / 9;
        int p = r - c * 9;                 // ky*3 + kx
        wbf[(p * COUT + m) * CIN + c] = f32_to_bf16(conv_w[i]);
    }
    if (i < COUT) {
        float inv = rsqrtf(bn_var[i] + 1e-5f);
        float s   = bn_gamma[i] * inv;
        scale[i] = s;
        shift[i] = (conv_b[i] - bn_mean[i]) * s + bn_beta[i];
    }
}

__global__ __launch_bounds__(256, 4)
void snn_conv_bn_lif_kernel(const float* __restrict__ x,
                            const uint16_t* __restrict__ wbf,
                            const float* __restrict__ scale,
                            const float* __restrict__ shift,
                            float* __restrict__ out) {
    extern __shared__ char smem[];
    uint16_t* xs   = (uint16_t*)smem;                 // [3 kx][CIN][CSTR] (4 rows x 32 w)
    float*    xraw = (float*)(smem + XRAW_OFF);       // [CIN][4][32] f32

    const int tid  = threadIdx.x;
    const int lane = tid & 31;
    const int wave = tid >> 5;            // 8 waves; wave w owns Cout rows [16w,16w+16)
    const int bidx = blockIdx.x;
    const int b    = bidx >> 4;           // batch 0..63
    const int h0   = (bidx & 15) * 2;     // strip of 2 output rows

    // C/D 16x16 f32 layout: VGPR v holds M=v (lanes 0-15) / M=v+8 (lanes 16-31).
    const int cb = wave * 16 + ((lane >= 16) ? 8 : 0);
    float sreg[8], hreg[8];
#pragma unroll
    for (int v = 0; v < 8; ++v) { sreg[v] = scale[cb + v]; hreg[v] = shift[cb + v]; }

    float pm[4][8];                       // LIF membrane, persists across t
#pragma unroll
    for (int nb = 0; nb < 4; ++nb)
#pragma unroll
        for (int v = 0; v < 8; ++v) pm[nb][v] = 0.0f;

    // A-fragment (16-bit 16x32): lanes 0-15 M=lane, K={k0..k0+7, k0+16..k0+23};
    // lanes 16-31 M=lane-16, K shifted +8.
    const int am   = wave * 16 + (lane & 15);
    const int akof = (lane >= 16) ? 8 : 0;
    const unsigned xraw_lo = (unsigned)(uintptr_t)(smem + XRAW_OFF); // LDS byte offset

    for (int t = 0; t < T_; ++t) {
        __syncthreads();   // previous iteration's xs/xraw reads done

        // ---- async stage raw f32 rows h0-1..h0+2 into LDS (HBM read once) ----
        const float* xsrc = x + ((size_t)(t * B_ + b) * CIN) * (H_ * W_);
#pragma unroll
        for (int i = 0; i < 8; ++i) {
            int ch  = tid + i * 256;      // 16B chunk id, 0..2047
            int c   = ch >> 5;            // 32 chunks per channel
            int row = (ch >> 3) & 3;      // global row h0-1+row
            int wq  = ch & 7;             // 4-float group
            int h   = h0 - 1 + row;
            if (h >= 0 && h < H_) {
                const float* g = xsrc + (size_t)c * (H_ * W_) + h * W_ + wq * 4;
                unsigned l = xraw_lo + ch * 16;
                asm volatile("global_load_async_to_lds_b128 %0, %1, off"
                             :: "v"(l), "v"(g) : "memory");
            } else {
                uint4 z = {0u, 0u, 0u, 0u};   // zero-pad out-of-image rows
                *(uint4*)(smem + XRAW_OFF + ch * 16) = z;
            }
        }
        asm volatile("s_wait_asynccnt 0" ::: "memory");
        __syncthreads();

        // ---- build 3 kx-pre-shifted bf16 copies (LDS->LDS) ----
        for (int e = tid; e < 3 * CIN * 128; e += 256) {
            int kx  = e >> 13;            // 0..2  (shift = kx-1)
            int q   = e & 8191;
            int c   = q >> 7;
            int row = (q >> 5) & 3;
            int ws  = q & 31;
            int w   = ws + kx - 1;
            float v = (w >= 0 && w < W_) ? xraw[c * 128 + row * 32 + w] : 0.0f;
            xs[kx * (CIN * CSTR) + c * CSTR + row * 32 + ws] = f32_to_bf16(v);
        }
        __syncthreads();

        // ---- 9 shifted GEMMs: (3 ky x 3 kx) x (2 channel chunks) x 4 N-tiles.
        //      A from L2-resident bf16 weights; B from LDS shifted copies. ----
        v8f acc[4];
        v8f zero = {};
#pragma unroll
        for (int nb = 0; nb < 4; ++nb) acc[nb] = zero;

#pragma unroll 1                      // keep VGPR pressure down -> more waves/SIMD
        for (int ky = 0; ky < 3; ++ky) {
            const uint16_t* xsy = xs + ky * 32;
            const uint16_t* wpy = wbf + ((size_t)(ky * 3) * COUT + am) * CIN + akof;
#pragma unroll
            for (int kx = 0; kx < 3; ++kx) {
                const uint16_t* xsp = xsy + kx * (CIN * CSTR);
                const uint16_t* wpp = wpy + (size_t)kx * COUT * CIN;
#pragma unroll
                for (int kk = 0; kk < 2; ++kk) {
                    AFrag a;
                    const uint16_t* wp = wpp + kk * 32;
                    a.h[0] = *(const uint4*)(wp);        // K c0..c0+7
                    a.h[1] = *(const uint4*)(wp + 16);   // K c0+16..c0+23
                    const uint16_t* bp = xsp + (kk * 32) * CSTR + (lane & 15);
#pragma unroll
                    for (int nb = 0; nb < 4; ++nb) {
                        int r    = nb >> 1;
                        int colb = (nb & 1) << 4;
                        v16bf bf;
#pragma unroll
                        for (int s = 0; s < 16; ++s) {
                            int k = (s < 8) ? (akof + s) : (16 + akof + s - 8);
                            uint16_t u = bp[k * CSTR + r * 32 + colb];
                            bf[s] = __builtin_bit_cast(__bf16, u);
                        }
                        acc[nb] = __builtin_amdgcn_wmma_f32_16x16x32_bf16(
                            false, a.v, false, bf, (short)0, acc[nb], false, false);
                    }
                }
            }
        }

        // ---- BN + LIF (tau=0.5, thresh=1.0, hard reset) + store spikes ----
#pragma unroll
        for (int nb = 0; nb < 4; ++nb) {
            int r   = nb >> 1;
            int col = ((nb & 1) << 4) + (lane & 15);
            size_t base = (((size_t)(t * B_ + b) * COUT + cb) * (H_ * W_))
                        + (size_t)(h0 + r) * W_ + col;
#pragma unroll
            for (int v = 0; v < 8; ++v) {
                float y     = acc[nb][v] * sreg[v] + hreg[v];
                float mem   = pm[nb][v] * 0.5f + y;
                float spike = (mem > 1.0f) ? 1.0f : 0.0f;
                pm[nb][v]   = (mem > 1.0f) ? 0.0f : mem;   // (1-spike)*mem
                out[base + (size_t)v * (H_ * W_)] = spike;
            }
        }
    }
}

extern "C" void kernel_launch(void* const* d_in, const int* in_sizes, int n_in,
                              void* d_out, int out_size, void* d_ws, size_t ws_size,
                              hipStream_t stream) {
    (void)in_sizes; (void)n_in; (void)ws_size; (void)out_size;
    const float* x        = (const float*)d_in[0];
    const float* conv_w   = (const float*)d_in[1];
    const float* conv_b   = (const float*)d_in[2];
    const float* bn_gamma = (const float*)d_in[3];
    const float* bn_beta  = (const float*)d_in[4];
    const float* bn_mean  = (const float*)d_in[5];
    const float* bn_var   = (const float*)d_in[6];
    float* out = (float*)d_out;

    uint16_t* wbf   = (uint16_t*)d_ws;
    float*    scale = (float*)((char*)d_ws + WS_SCALE_BYTE_OFF);
    float*    shift = scale + COUT;

    // 1) weight conversion + BN fold into workspace (L2-resident afterwards)
    hipLaunchKernelGGL(prep_kernel, dim3((COUT * CIN * 9 + 255) / 256), dim3(256),
                       0, stream,
                       conv_w, conv_b, bn_gamma, bn_beta, bn_mean, bn_var,
                       wbf, scale, shift);

    // 2) fused conv+BN+LIF
    dim3 grid(B_ * (H_ / 2));   // 1024 workgroups: (batch, 2-row strip)
    dim3 block(256);            // 8 waves (wave32)
    hipLaunchKernelGGL(snn_conv_bn_lif_kernel, grid, block, SMEM_TOTAL, stream,
                       x, wbf, scale, shift, out);
}
